// ExtractNet_6811818131895
// MI455X (gfx1250) — matrix operand order repaired and checked by reading, not compiled, not run on hardware
//
#include <hip/hip_runtime.h>

// ---------------------------------------------------------------------------
// MMoE extraction network for MI455X (gfx1250, wave32, WMMA f32_16x16x32_f16)
// B=65536, IN=256, H1=H2=OUT=64, T=2, ET=4, ES=4
// ---------------------------------------------------------------------------

#define NB    65536
#define DIN   256
#define DH    64
#define DOUT  64
#define NT    2
#define NET   4
#define NES   4

typedef __attribute__((ext_vector_type(16))) _Float16 v16h;
typedef __attribute__((ext_vector_type(8)))  _Float16 v8h;
typedef __attribute__((ext_vector_type(8)))  float    v8f;
typedef __attribute__((ext_vector_type(4)))  float    v4f;

__device__ __forceinline__ v8f wmma16(v16h a, v16h b, v8f c) {
  // D = A(16x32 f16) * B(32x16 f16) + C(16x16 f32)
  return __builtin_amdgcn_wmma_f32_16x16x32_f16(
      /*neg_a=*/false, a, /*neg_b=*/false, b,
      /*c_mod=*/(short)0, c, /*reuse_a=*/false, /*reuse_b=*/false);
}

// B-fragment for a 32x16 tile from an LDS weight matrix stored transposed as
// [N][K] f16.  ISA layout: lanes 0-15 hold K=k0..k0+7 and k0+16..k0+23 of
// column N=n0+lane, lanes 16-31 hold the same K shifted by +8 of column
// N=n0+lane-16.  Two 16B LDS loads per lane.
__device__ __forceinline__ v16h load_bfrag(const _Float16* WT, int Kdim,
                                           int n0, int k0, int lane) {
  const int n = n0 + (lane & 15);
  const int k = k0 + ((lane >> 4) << 3);
  const _Float16* p = WT + n * Kdim + k;
  v8h lo = *(const v8h*)p;
  v8h hi = *(const v8h*)(p + 16);
  v16h b;
#pragma unroll
  for (int j = 0; j < 8; ++j) { b[j] = lo[j]; b[8 + j] = hi[j]; }
  return b;
}

// A-fragment for a 16x32 tile from a per-wave LDS [16][64] f16 activation.
// ISA layout: lanes 0-15 (row=lane): K = base..base+7 and base+16..base+23;
// lanes 16-31 (row=lane-16): K shifted by +8.
__device__ __forceinline__ v16h load_afrag(const _Float16* H, int kt, int lane) {
  const int row  = lane & 15;
  const int kofs = (lane >> 4) << 3;  // 0 or 8
  const _Float16* p = H + row * 64 + kt * 32 + kofs;
  v8h lo = *(const v8h*)p;
  v8h hi = *(const v8h*)(p + 16);
  v16h a;
#pragma unroll
  for (int j = 0; j < 8; ++j) { a[j] = lo[j]; a[8 + j] = hi[j]; }
  return a;
}

// Cooperative load of a [K][N] f32 global weight matrix into LDS as
// transposed [N][K] f16.  N must be a multiple of 4.
__device__ __forceinline__ void load_wT(const float* W, _Float16* WT,
                                        int K, int N, int tid) {
  const int total = K * N;
  for (int i4 = tid * 4; i4 < total; i4 += 256 * 4) {
    v4f v = *(const v4f*)(W + i4);
    int k = i4 / N;
    int n = i4 % N;
#pragma unroll
    for (int m = 0; m < 4; ++m) WT[(n + m) * K + k] = (_Float16)v[m];
  }
}

// Prefetch an f32 weight matrix (count floats) into cache for the next expert
// iteration; lowers to global_prefetch_b8 (one cacheline per address).
__device__ __forceinline__ void prefetch_w(const float* W, int count, int tid) {
  for (int f = tid * 32; f < count; f += 256 * 32)
    __builtin_prefetch(W + f, 0, 1);
}

extern "C" __global__ __launch_bounds__(256)
void moe_extract_kernel(const float* __restrict__ X,
                        const float* __restrict__ Wt1, const float* __restrict__ bt1,
                        const float* __restrict__ Wt2, const float* __restrict__ bt2,
                        const float* __restrict__ Wt3, const float* __restrict__ bt3,
                        const float* __restrict__ Ws1, const float* __restrict__ bs1,
                        const float* __restrict__ Ws2, const float* __restrict__ bs2,
                        const float* __restrict__ Ws3, const float* __restrict__ bs3,
                        const float* __restrict__ Wg,  const float* __restrict__ bg,
                        float* __restrict__ Out) {
  // LDS map (f16 elements), total 64 KB:
  //   [0,16384)      W1^T [64][256]   (gate W^T [16][256] aliases this early)
  //   [16384,20480)  W2^T [64][64]
  //   [20480,24576)  W3^T [64][64]
  //   [24576,32768)  per-wave h staging: wave w at +w*1024, [16][64]
  __shared__ __align__(128) _Float16 lds[32768];
  _Float16* const sW1T = lds;
  _Float16* const sW2T = lds + 16384;
  _Float16* const sW3T = lds + 20480;

  const int tid  = threadIdx.x;
  const int lane = tid & 31;
  const int wave = tid >> 5;
  const int rowbase = blockIdx.x * 128 + wave * 16;
  _Float16* const sH = lds + 24576 + wave * 1024;

  const int colb   = lane & 15;
  const int rowoff = (lane >> 4) << 3;  // C-matrix row offset: 0 or 8

  // ---- X A-fragments: [16,256] f32 -> 8 f16 16x32 tiles in registers ----
  v16h xa[8];
  {
    const float* xr = X + (size_t)(rowbase + (lane & 15)) * DIN + ((lane >> 4) << 3);
#pragma unroll
    for (int kt = 0; kt < 8; ++kt) {
      const float* p = xr + kt * 32;
      v4f f0 = *(const v4f*)p;
      v4f f1 = *(const v4f*)(p + 4);
      v4f f2 = *(const v4f*)(p + 16);
      v4f f3 = *(const v4f*)(p + 20);
      v16h a;
#pragma unroll
      for (int j = 0; j < 4; ++j) {
        a[j]      = (_Float16)f0[j];
        a[4 + j]  = (_Float16)f1[j];
        a[8 + j]  = (_Float16)f2[j];
        a[12 + j] = (_Float16)f3[j];
      }
      xa[kt] = a;
    }
  }

  // ---- gate weights [T][256][8] -> LDS [16][256] f16 (col n = t*8+e) ----
  for (int i4 = tid * 4; i4 < NT * DIN * 8; i4 += 256 * 4) {
    v4f v = *(const v4f*)(Wg + i4);
    int t = i4 >> 11;
    int rem = i4 & 2047;
    int k = rem >> 3, e0 = rem & 7;
#pragma unroll
    for (int m = 0; m < 4; ++m) sW1T[(t * 8 + e0 + m) * DIN + k] = (_Float16)v[m];
  }
  __syncthreads();

  // ---- gate GEMM [16,256]x[256,16] + per-(row,task) softmax over 8 lanes ----
  v8f gw;
  {
    v8f c = {};
#pragma unroll
    for (int kt = 0; kt < 8; ++kt)
      c = wmma16(xa[kt], load_bfrag(sW1T, DIN, 0, kt * 32, lane), c);
    const float bv = bg[colb];
#pragma unroll
    for (int r = 0; r < 8; ++r) {
      float x = c[r] + bv;
      float m = x;
      m = fmaxf(m, __shfl_xor(m, 1, 32));
      m = fmaxf(m, __shfl_xor(m, 2, 32));
      m = fmaxf(m, __shfl_xor(m, 4, 32));
      float e = __expf(x - m);
      float s = e;
      s += __shfl_xor(s, 1, 32);
      s += __shfl_xor(s, 2, 32);
      s += __shfl_xor(s, 4, 32);
      gw[r] = e / s;
    }
  }

  v8f acc0[4] = {{}, {}, {}, {}};
  v8f acc1[4] = {{}, {}, {}, {}};

  // ---- expert loop: 8 task experts then 4 shared experts ----
#pragma unroll 1
  for (int idx = 0; idx < 12; ++idx) {
    const float *W1, *W2, *W3, *b1, *b2, *b3;
    int   c0 = 0, c1 = 0;
    float m0 = 0.f, m1 = 0.f;
    if (idx < 8) {
      const int t = idx >> 2, e = idx & 3;
      W1 = Wt1 + (size_t)idx * DIN * DH;  b1 = bt1 + idx * DH;
      W2 = Wt2 + (size_t)idx * DH * DH;   b2 = bt2 + idx * DH;
      W3 = Wt3 + (size_t)idx * DH * DOUT; b3 = bt3 + idx * DOUT;
      if (t == 0) { c0 = e;     m0 = 1.f; }
      else        { c1 = 8 + e; m1 = 1.f; }
    } else {
      const int s = idx - 8;
      W1 = Ws1 + (size_t)s * DIN * DH;  b1 = bs1 + s * DH;
      W2 = Ws2 + (size_t)s * DH * DH;   b2 = bs2 + s * DH;
      W3 = Ws3 + (size_t)s * DH * DOUT; b3 = bs3 + s * DOUT;
      c0 = 4 + s;  m0 = 1.f;
      c1 = 12 + s; m1 = 1.f;
    }

    __syncthreads();  // previous iteration done with all LDS weight regions
    load_wT(W1, sW1T, DIN, DH, tid);
    load_wT(W2, sW2T, DH, DH, tid);
    load_wT(W3, sW3T, DH, DOUT, tid);
    __syncthreads();

    // Prefetch next expert's weights into cache so the L2 fetch overlaps the
    // 48 WMMAs below (global_prefetch_b8; no counter, no VGPR writeback).
    if (idx + 1 < 12) {
      const int nx = idx + 1;
      const float *nW1, *nW2, *nW3;
      if (nx < 8) {
        nW1 = Wt1 + (size_t)nx * DIN * DH;
        nW2 = Wt2 + (size_t)nx * DH * DH;
        nW3 = Wt3 + (size_t)nx * DH * DOUT;
      } else {
        const int s = nx - 8;
        nW1 = Ws1 + (size_t)s * DIN * DH;
        nW2 = Ws2 + (size_t)s * DH * DH;
        nW3 = Ws3 + (size_t)s * DH * DOUT;
      }
      prefetch_w(nW1, DIN * DH, tid);
      prefetch_w(nW2, DH * DH, tid);
      prefetch_w(nW3, DH * DOUT, tid);
    }

    // layer 1: [16,256] x [256,64] -> relu -> sH (f16)
#pragma unroll
    for (int nt = 0; nt < 4; ++nt) {
      v8f c = {};
#pragma unroll
      for (int kt = 0; kt < 8; ++kt)
        c = wmma16(xa[kt], load_bfrag(sW1T, DIN, nt * 16, kt * 32, lane), c);
      const float bv = b1[nt * 16 + colb];
#pragma unroll
      for (int r = 0; r < 8; ++r)
        sH[(rowoff + r) * 64 + nt * 16 + colb] = (_Float16)fmaxf(c[r] + bv, 0.f);
    }

    // layer 2: [16,64] x [64,64] -> relu -> sH
    v16h ha0 = load_afrag(sH, 0, lane);
    v16h ha1 = load_afrag(sH, 1, lane);
#pragma unroll
    for (int nt = 0; nt < 4; ++nt) {
      v8f c = {};
      c = wmma16(ha0, load_bfrag(sW2T, DH, nt * 16, 0, lane), c);
      c = wmma16(ha1, load_bfrag(sW2T, DH, nt * 16, 32, lane), c);
      const float bv = b2[nt * 16 + colb];
#pragma unroll
      for (int r = 0; r < 8; ++r)
        sH[(rowoff + r) * 64 + nt * 16 + colb] = (_Float16)fmaxf(c[r] + bv, 0.f);
    }

    // layer 3: [16,64] x [64,64] + bias, then gated accumulate
    ha0 = load_afrag(sH, 0, lane);
    ha1 = load_afrag(sH, 1, lane);

    float g0r[8], g1r[8];
#pragma unroll
    for (int r = 0; r < 8; ++r) {
      const float gl0 = __shfl(gw[r], c0, 32);
      const float gh0 = __shfl(gw[r], c0 + 16, 32);
      const float gl1 = __shfl(gw[r], c1, 32);
      const float gh1 = __shfl(gw[r], c1 + 16, 32);
      g0r[r] = (lane < 16 ? gl0 : gh0) * m0;
      g1r[r] = (lane < 16 ? gl1 : gh1) * m1;
    }

#pragma unroll
    for (int nt = 0; nt < 4; ++nt) {
      v8f c = {};
      c = wmma16(ha0, load_bfrag(sW3T, DH, nt * 16, 0, lane), c);
      c = wmma16(ha1, load_bfrag(sW3T, DH, nt * 16, 32, lane), c);
      const float bv = b3[nt * 16 + colb];
#pragma unroll
      for (int r = 0; r < 8; ++r) {
        const float o = c[r] + bv;
        acc0[nt][r] += g0r[r] * o;
        acc1[nt][r] += g1r[r] * o;
      }
    }
  }

  // ---- store [B][T][64] f32 ----
#pragma unroll
  for (int nt = 0; nt < 4; ++nt) {
#pragma unroll
    for (int r = 0; r < 8; ++r) {
      const size_t row = (size_t)(rowbase + rowoff + r);
      Out[(row * NT + 0) * DOUT + nt * 16 + colb] = acc0[nt][r];
      Out[(row * NT + 1) * DOUT + nt * 16 + colb] = acc1[nt][r];
    }
  }
}

extern "C" void kernel_launch(void* const* d_in, const int* in_sizes, int n_in,
                              void* d_out, int out_size, void* d_ws, size_t ws_size,
                              hipStream_t stream) {
  const float* X   = (const float*)d_in[0];
  const float* Wt1 = (const float*)d_in[1];
  const float* bt1 = (const float*)d_in[2];
  const float* Wt2 = (const float*)d_in[3];
  const float* bt2 = (const float*)d_in[4];
  const float* Wt3 = (const float*)d_in[5];
  const float* bt3 = (const float*)d_in[6];
  const float* Ws1 = (const float*)d_in[7];
  const float* bs1 = (const float*)d_in[8];
  const float* Ws2 = (const float*)d_in[9];
  const float* bs2 = (const float*)d_in[10];
  const float* Ws3 = (const float*)d_in[11];
  const float* bs3 = (const float*)d_in[12];
  const float* Wg  = (const float*)d_in[13];
  const float* bg  = (const float*)d_in[14];
  float* Out = (float*)d_out;

  dim3 grid(NB / 128);
  dim3 block(256);
  moe_extract_kernel<<<grid, block, 0, stream>>>(
      X, Wt1, bt1, Wt2, bt2, Wt3, bt3,
      Ws1, bs1, Ws2, bs2, Ws3, bs3, Wg, bg, Out);
}
